// NeuralMemory_50044958933218
// MI455X (gfx1250) — matrix operand order repaired and checked by reading, not compiled, not run on hardware
//
#include <hip/hip_runtime.h>
#include <hip/hip_bf16.h>
#include <math.h>

#define B_ 16
#define S_ 4096
#define D_ 1024
#define M_ 512
#define CHUNK 64
#define NB 32
#define SSTRIDE 68   // padded row stride for the 64x64 score buffer (bank-conflict-free)

typedef __attribute__((ext_vector_type(2)))  float        v2f;
typedef __attribute__((ext_vector_type(8)))  float        v8f;
typedef __attribute__((ext_vector_type(16))) __bf16       v16bf;
typedef __attribute__((ext_vector_type(4)))  unsigned int v4u;
typedef __attribute__((ext_vector_type(8)))  int          v8i;
typedef __attribute__((ext_vector_type(4)))  int          v4i;

__device__ __forceinline__ __bf16 tobf(float x) { return (__bf16)x; }

// LDS aperture: flat addr[31:0] is the LDS byte offset (ISA 10.2 aperture table)
__device__ __forceinline__ unsigned lds_byte_off(const void* p) {
    return (unsigned)(uintptr_t)p;
}

// ---------------------------------------------------------------------------
// Tensor Data Mover: issue a 2D (or 1D when tile_d1==0) fp32 tile load
// Global -> LDS.  Descriptor bit layout per CDNA5 ISA 8.3/8.4 (D# group 0/1).
// data_size code 2 == 4 bytes.  Groups 2/3 + extra group zero (tensor <= 2D).
// This toolchain exposes the 6-arg builtin form.
// ---------------------------------------------------------------------------
__device__ __forceinline__ void tdm_load_f32(const void* gptr, unsigned lds_off,
                                             unsigned tile_d0, unsigned tile_d1,
                                             unsigned tensor_d0, unsigned tensor_d1,
                                             unsigned long long stride0_elems)
{
    unsigned long long ga = (unsigned long long)(uintptr_t)gptr;
    v4u g0;
    g0.x = 1u;                                   // count=1 (valid), user mode
    g0.y = lds_off;                              // lds_addr [63:32]
    g0.z = (unsigned)(ga & 0xffffffffu);         // global_addr [95:64]
    g0.w = (unsigned)((ga >> 32) & 0x01ffffffu)  // global_addr [120:96]
         | (2u << 30);                           // type=2 ("image") [127:126]
    v8i g1;
    g1[0] = (int)(2u << 16);                     // wg_mask=0, data_size=2 (4B)
    g1[1] = (int)((tensor_d0 & 0xffffu) << 16);  // tensor_dim0[15:0] @ [63:48]
    g1[2] = (int)(((tensor_d0 >> 16) & 0xffffu)  // tensor_dim0[31:16]
         | ((tensor_d1 & 0xffffu) << 16));       // tensor_dim1[15:0]
    g1[3] = (int)(((tensor_d1 >> 16) & 0xffffu)  // tensor_dim1[31:16]
         | ((tile_d0 & 0xffffu) << 16));         // tile_dim0 @ [127:112]
    g1[4] = (int)(tile_d1 & 0xffffu);            // tile_dim1; tile_dim2=0
    g1[5] = (int)(unsigned)(stride0_elems & 0xffffffffull);  // dim0_stride[31:0]
    g1[6] = (int)(unsigned)((stride0_elems >> 32) & 0xffffull); // dim0_stride[47:32]
    g1[7] = 0;                                   // dim1_stride = 0 (2D tile)
    v4i gz4 = {0, 0, 0, 0};
    v8i gz8 = {0, 0, 0, 0, 0, 0, 0, 0};
    __builtin_amdgcn_tensor_load_to_lds(g0, g1, gz4, gz4, gz8, 0);
}

// ---------------------------------------------------------------------------
// Kernel 1: fused QKV projection.  out[g][row, col] = sum_d x[row,d]*W[col,d] + b
// bf16 WMMA 16x16x32 with fp32 accumulation.
// grid = (B*S/16, 3), block = 256 (8 waves), each wave owns 64 output cols.
// ---------------------------------------------------------------------------
__global__ __launch_bounds__(256) void qkv_proj_kernel(
    const float* __restrict__ x,
    const float* __restrict__ Wq, const float* __restrict__ bq,
    const float* __restrict__ Wk, const float* __restrict__ bk,
    const float* __restrict__ Wv, const float* __restrict__ bv,
    float* __restrict__ qout, float* __restrict__ kout, float* __restrict__ vout)
{
    const int r0 = blockIdx.x * 16;     // row tile in flattened (B*S)
    const int g  = blockIdx.y;          // 0=q, 1=k, 2=v
    const float* W; const float* bias; float* outp;
    if (g == 0)      { W = Wq; bias = bq; outp = qout; }
    else if (g == 1) { W = Wk; bias = bk; outp = kout; }
    else             { W = Wv; bias = bv; outp = vout; }

    const int lane = threadIdx.x & 31;
    const int wave = threadIdx.x >> 5;
    const int n0   = wave * 64;
    const int hi   = (lane >> 4) & 1;
    const int l15  = lane & 15;

    const float* xrow = x + (size_t)(r0 + l15) * D_;  // A-frag row (M=lane%16)
    const int abase = hi ? 16 : 0;   // A layout: lanes<16 K=0..15, lanes>=16 K=16..31
    const int bbase = hi ? 16 : 0;   // B 32x16 layout: lanes<16 K=0..15, lanes>=16 K=16..31

    v8f acc[4] = {};

    for (int kk = 0; kk < D_; kk += 32) {
        __builtin_prefetch(xrow + kk + 64, 0, 1);   // global_prefetch_b8
        v16bf a;
        #pragma unroll
        for (int i = 0; i < 4; ++i) {
            v2f p0 = *(const v2f*)(xrow + kk + abase + 2 * i);
            v2f p1 = *(const v2f*)(xrow + kk + 8 + abase + 2 * i);
            a[2*i]     = tobf(p0.x); a[2*i + 1]     = tobf(p0.y);
            a[8 + 2*i] = tobf(p1.x); a[8 + 2*i + 1] = tobf(p1.y);
        }
        #pragma unroll
        for (int j = 0; j < 4; ++j) {
            const float* wrow = W + (size_t)(n0 + j * 16 + l15) * D_ + kk + bbase;
            v16bf b;
            #pragma unroll
            for (int i = 0; i < 8; ++i) {
                v2f p = *(const v2f*)(wrow + 2 * i);
                b[2*i] = tobf(p.x); b[2*i + 1] = tobf(p.y);
            }
            acc[j] = __builtin_amdgcn_wmma_f32_16x16x32_bf16(
                false, a, false, b, (short)0, acc[j], false, false);
        }
    }
    #pragma unroll
    for (int j = 0; j < 4; ++j) {
        int col = n0 + j * 16 + l15;
        float bb = bias[col];
        #pragma unroll
        for (int r = 0; r < 8; ++r) {
            int row = r0 + r + hi * 8;
            outp[(size_t)row * M_ + col] = acc[j][r] + bb;
        }
    }
}

// ---------------------------------------------------------------------------
// Kernel 2: chunked linear-recurrence scan.  One block owns a (M x NB) fp32
// slice of Mem for one batch, kept in LDS for all S steps.  fp32 WMMA 16x16x4.
// K chunk (64x512) and V slice (64x32) are DMA'd into LDS with the TDM each
// chunk; fragments then come from LDS instead of 2KB-strided global loads.
// grid = (M/NB, B), block = 256 (8 waves).
// ---------------------------------------------------------------------------
__global__ __launch_bounds__(256) void scan_kernel(
    const float* __restrict__ qp, const float* __restrict__ kp,
    const float* __restrict__ vp, const float* __restrict__ alpha_p,
    float* __restrict__ out, float* __restrict__ memf)
{
    __shared__ __align__(16) float memS[M_ * NB];          //  64 KB state slice
    __shared__ __align__(16) float kS[CHUNK * M_];         // 128 KB K chunk (TDM)
    __shared__ __align__(16) float vS[CHUNK * NB];         //   8 KB V slice (TDM)
    __shared__ __align__(16) float Sbuf[CHUNK * SSTRIDE];  //  17 KB intra scores
    __shared__ float lamPow[CHUNK + 1];

    const int b    = blockIdx.y;
    const int n0   = blockIdx.x * NB;
    const int tid  = threadIdx.x;
    const int lane = tid & 31;
    const int wave = tid >> 5;
    const int hi   = (lane >> 4) & 1;
    const int l15  = lane & 15;
    const int koff = hi ? 2 : 0;      // f32 WMMA A/B: lanes<16 hold K0,K1; lanes>=16 K2,K3

    const float alpha = alpha_p[0];
    const float aG  = 1.f / (1.f + __expf(-alpha));
    const float lam = 1.f - aG;

    for (int i = tid; i < M_ * NB; i += 256) memS[i] = 0.f;
    if (tid <= CHUNK) lamPow[tid] = powf(lam, (float)tid);
    __syncthreads();

    const float* qB = qp + (size_t)b * S_ * M_;
    const float* kB = kp + (size_t)b * S_ * M_;
    const float* vB = vp + (size_t)b * S_ * M_;
    float* outB = out + (size_t)b * S_ * M_;

    const unsigned kS_off = lds_byte_off(kS);
    const unsigned vS_off = lds_byte_off(vS);

    for (int t0 = 0; t0 < S_; t0 += CHUNK) {
        const float* qc = qB + (size_t)t0 * M_;
        const float* kc = kB + (size_t)t0 * M_;
        const float* vc = vB + (size_t)t0 * M_;
        float* oc = outB + (size_t)t0 * M_;

        // ---- TDM: K chunk (1D contiguous 64*512 f32) and V slice (2D 32x64) -> LDS
        if (wave == 0) {
            tdm_load_f32(kc, kS_off, CHUNK * M_, 0, CHUNK * M_, 1, CHUNK * M_);
            tdm_load_f32(vc + n0, vS_off, NB, CHUNK, M_, CHUNK, M_);
            __builtin_amdgcn_s_wait_tensorcnt(0);
        }
        __syncthreads();   // publish TDM writes to all waves

        // ---- Step A: intra scores S[tau,s] = a*lam^(tau-1-s) * (q_tau . k_s), s<tau
        #pragma unroll
        for (int tt = 0; tt < 2; ++tt) {
            int tile = wave + tt * 8;          // 16 tiles of the 64x64 score matrix
            int tr = tile >> 2, tc = tile & 3;
            v8f acc = {};
            if (tc <= tr) {                    // strictly-upper tiles are all zero
                const float* qrow = qc + (size_t)(tr * 16 + l15) * M_;
                const float* krow = kS + (size_t)(tc * 16 + l15) * M_;
                for (int kk = 0; kk < M_; kk += 4) {
                    v2f a  = *(const v2f*)(qrow + kk + koff);
                    v2f bf = *(const v2f*)(krow + kk + koff);
                    acc = __builtin_amdgcn_wmma_f32_16x16x4_f32(
                        false, a, false, bf, (short)0, acc, false, false);
                }
            }
            #pragma unroll
            for (int r = 0; r < 8; ++r) {
                int tau = tr * 16 + r + hi * 8;
                int s   = tc * 16 + l15;
                float v = (s < tau) ? acc[r] * aG * lamPow[tau - 1 - s] : 0.f;
                Sbuf[tau * SSTRIDE + s] = v;
            }
        }
        __syncthreads();

        // ---- Step B: r = (lam^tau * Q) @ Mem + S @ V    (Mem + V in LDS)
        {
            int tr = wave >> 1;                // tau tile 0..3
            int tn = wave & 1;                 // n tile 0..1
            v8f acc = {};
            int tau_l = tr * 16 + l15;
            const float* qrow = qc + (size_t)tau_l * M_;
            float lscale = lamPow[tau_l];
            int ncol = tn * 16 + l15;
            for (int kk = 0; kk < M_; kk += 4) {
                v2f a = *(const v2f*)(qrow + kk + koff);
                a.x *= lscale; a.y *= lscale;
                v2f bf;
                bf.x = memS[(kk + koff)     * NB + ncol];
                bf.y = memS[(kk + koff + 1) * NB + ncol];
                acc = __builtin_amdgcn_wmma_f32_16x16x4_f32(
                    false, a, false, bf, (short)0, acc, false, false);
            }
            const float* srow = Sbuf + tau_l * SSTRIDE;
            for (int ss = 0; ss < CHUNK; ss += 4) {
                v2f a = *(const v2f*)(srow + ss + koff);
                v2f bf;
                bf.x = vS[(ss + koff)     * NB + ncol];
                bf.y = vS[(ss + koff + 1) * NB + ncol];
                acc = __builtin_amdgcn_wmma_f32_16x16x4_f32(
                    false, a, false, bf, (short)0, acc, false, false);
            }
            #pragma unroll
            for (int r = 0; r < 8; ++r) {
                int tau = tr * 16 + r + hi * 8;
                oc[(size_t)tau * M_ + n0 + tn * 16 + l15] = acc[r];
            }
        }
        __syncthreads();

        // ---- Step C: Mem = lam^C * Mem + (a*lam^(C-1-tau) * K)^T @ V   (K,V in LDS)
        {
            float lamC = lamPow[CHUNK];
            #pragma unroll
            for (int tt = 0; tt < 8; ++tt) {
                int tile = wave * 8 + tt;      // 64 tiles of the 512x32 slice
                int mr = tile >> 1;
                int tn = tile & 1;
                v8f acc = {};
                int m = mr * 16 + l15;
                int ncol = tn * 16 + l15;
                for (int ss = 0; ss < CHUNK; ss += 4) {
                    int s0 = ss + koff;
                    v2f a;
                    a.x = kS[(s0)     * M_ + m] * (aG * lamPow[CHUNK - 1 - s0]);
                    a.y = kS[(s0 + 1) * M_ + m] * (aG * lamPow[CHUNK - 2 - s0]);
                    v2f bf;
                    bf.x = vS[(s0)     * NB + ncol];
                    bf.y = vS[(s0 + 1) * NB + ncol];
                    acc = __builtin_amdgcn_wmma_f32_16x16x4_f32(
                        false, a, false, bf, (short)0, acc, false, false);
                }
                #pragma unroll
                for (int r = 0; r < 8; ++r) {
                    int mrow = mr * 16 + r + hi * 8;
                    int idx  = mrow * NB + tn * 16 + l15;
                    memS[idx] = lamC * memS[idx] + acc[r];
                }
            }
        }
        __syncthreads();   // end of chunk: kS/vS free for next TDM
    }

    // final Mem slice -> Mf output
    float* mf = memf + (size_t)b * M_ * M_;
    for (int i = tid; i < M_ * NB; i += 256) {
        int m = i / NB, n = i % NB;
        mf[(size_t)m * M_ + n0 + n] = memS[i];
    }
}

// ---------------------------------------------------------------------------
extern "C" void kernel_launch(void* const* d_in, const int* in_sizes, int n_in,
                              void* d_out, int out_size, void* d_ws, size_t ws_size,
                              hipStream_t stream) {
    const float* x     = (const float*)d_in[0];
    const float* Wq    = (const float*)d_in[1];
    const float* bq    = (const float*)d_in[2];
    const float* Wk    = (const float*)d_in[3];
    const float* bk    = (const float*)d_in[4];
    const float* Wv    = (const float*)d_in[5];
    const float* bv    = (const float*)d_in[6];
    const float* alpha = (const float*)d_in[7];

    float* out  = (float*)d_out;
    float* memf = out + (size_t)B_ * S_ * M_;

    float* q = (float*)d_ws;                       // 3 x (B,S,M) fp32 scratch
    float* k = q + (size_t)B_ * S_ * M_;
    float* v = k + (size_t)B_ * S_ * M_;

    dim3 g1((B_ * S_) / 16, 3);
    qkv_proj_kernel<<<g1, 256, 0, stream>>>(x, Wq, bq, Wk, bk, Wv, bv, q, k, v);

    dim3 g2(M_ / NB, B_);
    scan_kernel<<<g2, 256, 0, stream>>>(q, k, v, alpha, out, memf);
}
